// LinearTransformerBlock_20684562497541
// MI455X (gfx1250) — matrix operand order repaired and checked by reading, not compiled, not run on hardware
//
#include <hip/hip_runtime.h>
#include <hip/hip_bf16.h>

typedef unsigned short u16;
typedef unsigned int   u32;

typedef __bf16 bf16x16 __attribute__((ext_vector_type(16)));
typedef float  f32x8   __attribute__((ext_vector_type(8)));

union Frag16 { uint4 q[2]; bf16x16 v; };

constexpr int TOK = 65536;          // B*N tokens
constexpr int NEL = TOK * 256;      // elements of a (TOK,256) tensor

__device__ __forceinline__ u16 f2bf(float f) {
  u32 u = __builtin_bit_cast(u32, f);
  u32 r = u + 0x7FFFu + ((u >> 16) & 1u);
  return (u16)(r >> 16);
}
__device__ __forceinline__ float bf2f(u16 h) {
  u32 u = ((u32)h) << 16;
  return __builtin_bit_cast(float, u);
}
__device__ __forceinline__ f32x8 zero8() {
  f32x8 z;
#pragma unroll
  for (int i = 0; i < 8; ++i) z[i] = 0.f;
  return z;
}
__device__ __forceinline__ f32x8 wmma_bf16(const Frag16& a, const Frag16& b, f32x8 c) {
  return __builtin_amdgcn_wmma_f32_16x16x32_bf16(false, a.v, false, b.v, (short)0, c, false, false);
}

// ---------------------------------------------------------------------------
// bf16 GEMM, compile-time epilogue.
//   C[M,N] = act( (A@Bw [+ A2@B2] + bias) * oscale )
// FLAGS: 2=relu, 4=write f32 Cf, 8=write bf16 Cb, 16=bias
// DUAL : stream a second (A2,B2) pair through the K loop (effective 2K)
// Tiles 128x128x32, 256 threads = 8 waves (4x2), wave owns 32x64.
// Register double-buffered staging + global_prefetch for tile k+2.
// ---------------------------------------------------------------------------
#define LDP 40  // padded LDS row stride in ushorts (80B: 16B-aligned, conflict-free)

template <int FLAGS, bool DUAL>
__global__ __launch_bounds__(256) void gemm_bf16_kernel(
    const u16* __restrict__ A, const u16* __restrict__ Bw,
    const u16* __restrict__ A2, const u16* __restrict__ B2,
    const float* __restrict__ bias, float* Cf, u16* Cb,
    int M, int N, int K, float oscale) {
  __shared__ __align__(16) u16 As[128 * LDP];
  __shared__ __align__(16) u16 Bt[128 * LDP];

  int tid = threadIdx.x;
  int wave = tid >> 5, lane = tid & 31;
  int lan16 = lane & 15, khalf = lane >> 4;
  int wr = wave >> 1, wc = wave & 1;  // 4x2 wave grid
  int m0 = blockIdx.y * 128, n0 = blockIdx.x * 128;

  int arow = tid >> 1, aseg = (tid & 1) * 16;
  int bkr = tid >> 3, bnseg = (tid & 7) * 16;

  const int KT = DUAL ? 2 * K : K;

  f32x8 acc[2][4];
#pragma unroll
  for (int i = 0; i < 2; ++i)
#pragma unroll
    for (int j = 0; j < 4; ++j) acc[i][j] = zero8();

  uint4 aR[2], bR[2];
  auto load_tile = [&](int kk) {
    const u16* Ap = A;
    const u16* Bp = Bw;
    int k0 = kk;
    if (DUAL && kk >= K) { Ap = A2; Bp = B2; k0 = kk - K; }
    const u16* srcA = Ap + (size_t)(m0 + arow) * K + k0 + aseg;
    aR[0] = *(const uint4*)(srcA);
    aR[1] = *(const uint4*)(srcA + 8);
    const u16* srcB = Bp + (size_t)(k0 + bkr) * N + n0 + bnseg;
    bR[0] = *(const uint4*)(srcB);
    bR[1] = *(const uint4*)(srcB + 8);
  };
  auto prefetch_tile = [&](int kk) {
    const u16* Ap = A;
    const u16* Bp = Bw;
    int k0 = kk;
    if (DUAL && kk >= K) { Ap = A2; Bp = B2; k0 = kk - K; }
    __builtin_prefetch(Ap + (size_t)(m0 + arow) * K + k0 + aseg, 0, 3);
    __builtin_prefetch(Bp + (size_t)(k0 + bkr) * N + n0 + bnseg, 0, 3);
  };

  load_tile(0);
  for (int kk = 0; kk < KT; kk += 32) {
    // commit staged registers to LDS (B transposed -> Bt[n][k])
    *(uint4*)(&As[arow * LDP + aseg])     = aR[0];
    *(uint4*)(&As[arow * LDP + aseg + 8]) = aR[1];
    {
      u16 tmp[16];
      *(uint4*)(tmp)     = bR[0];
      *(uint4*)(tmp + 8) = bR[1];
#pragma unroll
      for (int e = 0; e < 16; ++e) Bt[(bnseg + e) * LDP + bkr] = tmp[e];
    }
    __syncthreads();

    if (kk + 32 < KT) load_tile(kk + 32);       // overlap with WMMA below
    if (kk + 64 < KT) prefetch_tile(kk + 64);   // global_prefetch_b8

    Frag16 aF[2], bF[4];
#pragma unroll
    for (int tm = 0; tm < 2; ++tm) {
      int r = wr * 32 + tm * 16 + lan16;
      aF[tm].q[0] = *(const uint4*)(&As[r * LDP + 8 * khalf]);
      aF[tm].q[1] = *(const uint4*)(&As[r * LDP + 16 + 8 * khalf]);
    }
#pragma unroll
    for (int tn = 0; tn < 4; ++tn) {
      int c = wc * 64 + tn * 16 + lan16;
      bF[tn].q[0] = *(const uint4*)(&Bt[c * LDP + 8 * khalf]);
      bF[tn].q[1] = *(const uint4*)(&Bt[c * LDP + 16 + 8 * khalf]);
    }
#pragma unroll
    for (int tm = 0; tm < 2; ++tm)
#pragma unroll
      for (int tn = 0; tn < 4; ++tn)
        acc[tm][tn] = wmma_bf16(aF[tm], bF[tn], acc[tm][tn]);
    __syncthreads();
  }

#pragma unroll
  for (int tm = 0; tm < 2; ++tm)
#pragma unroll
    for (int tn = 0; tn < 4; ++tn)
#pragma unroll
      for (int r = 0; r < 8; ++r) {
        int row = m0 + wr * 32 + tm * 16 + 8 * khalf + r;
        int col = n0 + wc * 64 + tn * 16 + lan16;
        float v = acc[tm][tn][r];
        if constexpr ((FLAGS & 16) != 0) v += bias[col];
        v *= oscale;
        if constexpr ((FLAGS & 2) != 0) v = v > 0.f ? v : 0.f;
        size_t idx = (size_t)row * N + col;
        if constexpr ((FLAGS & 4) != 0) Cf[idx] = v;
        if constexpr ((FLAGS & 8) != 0) Cb[idx] = f2bf(v);
      }
}

// ---------------------------------------------------------------------------
// LayerNorm over D=256, one workgroup per token, bf16 output.
// ---------------------------------------------------------------------------
__global__ __launch_bounds__(256) void ln_kernel(
    const float* __restrict__ x, const float* __restrict__ g,
    const float* __restrict__ b, u16* __restrict__ out) {
  __shared__ float red[256];
  int t = threadIdx.x;
  size_t row = blockIdx.x;
  float v = x[row * 256 + t];
  red[t] = v;
  __syncthreads();
  for (int s = 128; s > 0; s >>= 1) {
    if (t < s) red[t] += red[t + s];
    __syncthreads();
  }
  float mu = red[0] * (1.f / 256.f);
  __syncthreads();
  float d = v - mu;
  red[t] = d * d;
  __syncthreads();
  for (int s = 128; s > 0; s >>= 1) {
    if (t < s) red[t] += red[t + s];
    __syncthreads();
  }
  float var = red[0] * (1.f / 256.f);
  float y = g[t] * d * rsqrtf(var + 1e-3f) + b[t];
  out[row * 256 + t] = f2bf(y);
}

// ---------------------------------------------------------------------------
// phi: per-head Performer features.  Input Qs is (TOK,256) bf16 PRE-SCALED by
// DK^-0.25 (baked into the QKV GEMM oscale).  proj = x @ omega (K=32 WMMA),
// pq = exp(proj - 0.5*||x||^2) / sqrt(M).  Output layout (B,H,N,M).
// grid: (TOK/128, H)
// ---------------------------------------------------------------------------
__global__ __launch_bounds__(256) void phi_kernel(
    const u16* __restrict__ Qs, const float* __restrict__ omega,
    u16* __restrict__ outP) {
  __shared__ __align__(16) u16 As[128 * LDP];
  __shared__ __align__(16) u16 Bt[64 * LDP];
  __shared__ float sq[128];
  int t = threadIdx.x;
  int h = blockIdx.y;
  int tok0 = blockIdx.x * 128;
  {
    int row = t >> 1, seg = (t & 1) * 16;
    const u16* src = Qs + (size_t)(tok0 + row) * 256 + h * 32 + seg;
    *(uint4*)(&As[row * LDP + seg])     = *(const uint4*)(src);
    *(uint4*)(&As[row * LDP + seg + 8]) = *(const uint4*)(src + 8);
  }
  {
    int k = t >> 3, mseg = (t & 7) * 8;  // omega is 32x64 f32
#pragma unroll
    for (int e = 0; e < 8; ++e)
      Bt[(mseg + e) * LDP + k] = f2bf(omega[k * 64 + mseg + e]);
  }
  __syncthreads();
  if (t < 128) {
    float ss = 0.f;
#pragma unroll 8
    for (int kk = 0; kk < 32; ++kk) { float xv = bf2f(As[t * LDP + kk]); ss += xv * xv; }
    sq[t] = 0.5f * ss;
  }
  __syncthreads();
  int wave = t >> 5, lane = t & 31, lan16 = lane & 15, khalf = lane >> 4;
  int wr = wave >> 1, wc = wave & 1;
  f32x8 acc[2][2];
#pragma unroll
  for (int i = 0; i < 2; ++i)
#pragma unroll
    for (int j = 0; j < 2; ++j) acc[i][j] = zero8();
  Frag16 aF[2], bF[2];
#pragma unroll
  for (int tm = 0; tm < 2; ++tm) {
    int r = 32 * wr + 16 * tm + lan16;
    aF[tm].q[0] = *(const uint4*)(&As[r * LDP + 8 * khalf]);
    aF[tm].q[1] = *(const uint4*)(&As[r * LDP + 16 + 8 * khalf]);
  }
#pragma unroll
  for (int tn = 0; tn < 2; ++tn) {
    int c = 32 * wc + 16 * tn + lan16;
    bF[tn].q[0] = *(const uint4*)(&Bt[c * LDP + 8 * khalf]);
    bF[tn].q[1] = *(const uint4*)(&Bt[c * LDP + 16 + 8 * khalf]);
  }
#pragma unroll
  for (int tm = 0; tm < 2; ++tm)
#pragma unroll
    for (int tn = 0; tn < 2; ++tn)
      acc[tm][tn] = wmma_bf16(aF[tm], bF[tn], acc[tm][tn]);

  int b = tok0 >> 13;
  int nbase = tok0 & 8191;
  size_t obase = (size_t)(b * 8 + h) * 8192;
#pragma unroll
  for (int tm = 0; tm < 2; ++tm)
#pragma unroll
    for (int tn = 0; tn < 2; ++tn)
#pragma unroll
      for (int r = 0; r < 8; ++r) {
        int rr = 32 * wr + 16 * tm + 8 * khalf + r;
        int m = 32 * wc + 16 * tn + lan16;
        float v = expf(acc[tm][tn][r] - sq[rr]) * 0.125f;  // 1/sqrt(M=64)
        outP[(obase + nbase + rr) * 64 + m] = f2bf(v);
      }
}

// ---------------------------------------------------------------------------
// kv[bh] = pk^T @ v  (64x32, K=N=8192), plus s[bh][m] = sum_n pk[n][m].
// One workgroup per (b,h); 8 waves each own one 16x16 tile of the 64x32 output.
// ---------------------------------------------------------------------------
__global__ __launch_bounds__(256) void kv_kernel(
    const u16* __restrict__ pk, const u16* __restrict__ vmat,
    float* __restrict__ kvb, float* __restrict__ sb) {
  __shared__ __align__(16) u16 At[64 * 136];  // pk^T: At[m][n]
  __shared__ __align__(16) u16 Bt[32 * 136];  // v^T : Bt[d][n]
  int t = threadIdx.x;
  int bh = blockIdx.x;
  int b = bh >> 3, h = bh & 7;
  int wave = t >> 5, lane = t & 31, lan16 = lane & 15, khalf = lane >> 4;
  int wr = wave >> 1, wc = wave & 1;
  f32x8 acc = zero8();
  float sreg = 0.f;
  size_t pkBase = (size_t)bh * 8192 * 64;
  size_t vBase = (size_t)b * 8192 * 256 + h * 32;

  for (int n0 = 0; n0 < 8192; n0 += 128) {
    {
      int nr = t >> 1, mseg = (t & 1) * 32;
      const u16* src = pk + pkBase + (size_t)(n0 + nr) * 64 + mseg;
      u16 tmp[32];
      *(uint4*)(tmp)      = *(const uint4*)(src);
      *(uint4*)(tmp + 8)  = *(const uint4*)(src + 8);
      *(uint4*)(tmp + 16) = *(const uint4*)(src + 16);
      *(uint4*)(tmp + 24) = *(const uint4*)(src + 24);
      if (n0 + 128 < 8192) __builtin_prefetch(src + 128 * 64, 0, 3);
#pragma unroll
      for (int e = 0; e < 32; ++e) At[(mseg + e) * 136 + nr] = tmp[e];
    }
    {
      int nr = t >> 1, dseg = (t & 1) * 16;
      const u16* src = vmat + vBase + (size_t)(n0 + nr) * 256 + dseg;
      u16 tmp[16];
      *(uint4*)(tmp)     = *(const uint4*)(src);
      *(uint4*)(tmp + 8) = *(const uint4*)(src + 8);
      if (n0 + 128 < 8192) __builtin_prefetch(src + 128 * 256, 0, 3);
#pragma unroll
      for (int e = 0; e < 16; ++e) Bt[(dseg + e) * 136 + nr] = tmp[e];
    }
    __syncthreads();
    if (t < 64) {
#pragma unroll 8
      for (int nn = 0; nn < 128; ++nn) sreg += bf2f(At[t * 136 + nn]);
    }
    Frag16 aF, bF;
#pragma unroll
    for (int kn0 = 0; kn0 < 128; kn0 += 32) {
      int m = 16 * wr + lan16;
      aF.q[0] = *(const uint4*)(&At[m * 136 + kn0 + 8 * khalf]);
      aF.q[1] = *(const uint4*)(&At[m * 136 + kn0 + 16 + 8 * khalf]);
      int d = 16 * wc + lan16;
      bF.q[0] = *(const uint4*)(&Bt[d * 136 + kn0 + 8 * khalf]);
      bF.q[1] = *(const uint4*)(&Bt[d * 136 + kn0 + 16 + 8 * khalf]);
      acc = wmma_bf16(aF, bF, acc);
    }
    __syncthreads();
  }
#pragma unroll
  for (int r = 0; r < 8; ++r) {
    int m = 16 * wr + 8 * khalf + r;
    int d = 16 * wc + lan16;
    kvb[(size_t)bh * 2048 + m * 32 + d] = acc[r];
  }
  if (t < 64) sb[bh * 64 + t] = sreg;
}

// ---------------------------------------------------------------------------
// num = pq @ kv (K=64), den = pq . s, out = num/den -> attn (TOK,256) bf16.
// grid: (N/128, B*H)
// ---------------------------------------------------------------------------
__global__ __launch_bounds__(256) void num_kernel(
    const u16* __restrict__ pq, const float* __restrict__ kvb,
    const float* __restrict__ sb, u16* __restrict__ attn) {
  __shared__ __align__(16) u16 As[128 * 72];  // pq tile
  __shared__ __align__(16) u16 Bt[32 * 72];   // kv^T: Bt[d][m]
  __shared__ float sL[64];
  __shared__ float den[128];
  int t = threadIdx.x;
  int bh = blockIdx.y;
  int b = bh >> 3, h = bh & 7;
  int n0 = blockIdx.x * 128;
  size_t pqBase = (size_t)bh * 8192 * 64;
  {
    int row = t >> 1, seg = (t & 1) * 32;
    const u16* src = pq + pqBase + (size_t)(n0 + row) * 64 + seg;
    *(uint4*)(&As[row * 72 + seg])      = *(const uint4*)(src);
    *(uint4*)(&As[row * 72 + seg + 8])  = *(const uint4*)(src + 8);
    *(uint4*)(&As[row * 72 + seg + 16]) = *(const uint4*)(src + 16);
    *(uint4*)(&As[row * 72 + seg + 24]) = *(const uint4*)(src + 24);
  }
  for (int i = t; i < 2048; i += 256) {
    int m = i >> 5, d = i & 31;
    Bt[d * 72 + m] = f2bf(kvb[(size_t)bh * 2048 + i]);
  }
  if (t < 64) sL[t] = sb[bh * 64 + t];
  __syncthreads();
  if (t < 128) {
    float a = 0.f;
#pragma unroll 8
    for (int m = 0; m < 64; ++m) a += bf2f(As[t * 72 + m]) * sL[m];
    den[t] = a;
  }
  __syncthreads();
  int wave = t >> 5, lane = t & 31, lan16 = lane & 15, khalf = lane >> 4;
  f32x8 acc[2];
  acc[0] = zero8(); acc[1] = zero8();
#pragma unroll
  for (int k0 = 0; k0 < 64; k0 += 32) {
    Frag16 aF;
    int row = 16 * wave + lan16;
    aF.q[0] = *(const uint4*)(&As[row * 72 + k0 + 8 * khalf]);
    aF.q[1] = *(const uint4*)(&As[row * 72 + k0 + 16 + 8 * khalf]);
#pragma unroll
    for (int tn = 0; tn < 2; ++tn) {
      Frag16 bF;
      int d = 16 * tn + lan16;
      bF.q[0] = *(const uint4*)(&Bt[d * 72 + k0 + 8 * khalf]);
      bF.q[1] = *(const uint4*)(&Bt[d * 72 + k0 + 16 + 8 * khalf]);
      acc[tn] = wmma_bf16(aF, bF, acc[tn]);
    }
  }
  size_t tokBase = (size_t)b * 8192 + n0;
#pragma unroll
  for (int tn = 0; tn < 2; ++tn)
#pragma unroll
    for (int r = 0; r < 8; ++r) {
      int rr = 16 * wave + 8 * khalf + r;
      int d = 16 * tn + lan16;
      float v = acc[tn][r] / den[rr];
      attn[(tokBase + rr) * 256 + h * 32 + d] = f2bf(v);
    }
}

// ---------------------------------------------------------------------------
// Elementwise helpers.
// ---------------------------------------------------------------------------
__global__ __launch_bounds__(256) void cast_kernel(const float* __restrict__ s,
                                                   u16* __restrict__ d, int n) {
  int i = blockIdx.x * 256 + threadIdx.x;
  if (i < n) d[i] = f2bf(s[i]);
}

__global__ __launch_bounds__(256) void rx_kernel(const float* __restrict__ R,
                                                 const float* __restrict__ x,
                                                 u16* __restrict__ rx, int n) {
  int i = blockIdx.x * 256 + threadIdx.x;
  if (i >= n) return;
  float r = 1.f / (1.f + expf(-R[i]));
  rx[i] = f2bf(r * x[i]);
}

__global__ __launch_bounds__(256) void gate_combine_kernel(
    const float* __restrict__ Z, const float* __restrict__ G,
    const float* __restrict__ x, const float* __restrict__ bg,
    float* __restrict__ oF, u16* __restrict__ oB, int writeB, int n) {
  int i = blockIdx.x * 256 + threadIdx.x;
  if (i >= n) return;
  int col = i & 255;
  float z = 1.f / (1.f + expf(-(Z[i] - bg[col])));
  float hh = tanhf(G[i]);
  float o = (1.f - z) * x[i] + z * hh;
  oF[i] = o;
  if (writeB) oB[i] = f2bf(o);
}

// ---------------------------------------------------------------------------
// Host orchestration.
// ---------------------------------------------------------------------------
extern "C" void kernel_launch(void* const* d_in, const int* in_sizes, int n_in,
                              void* d_out, int out_size, void* d_ws, size_t ws_size,
                              hipStream_t stream) {
  (void)in_sizes; (void)n_in; (void)out_size; (void)ws_size;
  const float* x     = (const float*)d_in[0];
  const float* ln1g  = (const float*)d_in[1];
  const float* ln1b  = (const float*)d_in[2];
  const float* Wq    = (const float*)d_in[3];
  const float* bq    = (const float*)d_in[4];
  const float* Wk    = (const float*)d_in[5];
  const float* bk    = (const float*)d_in[6];
  const float* Wv    = (const float*)d_in[7];
  const float* bv    = (const float*)d_in[8];
  const float* Wo    = (const float*)d_in[9];
  const float* bo    = (const float*)d_in[10];
  const float* omega = (const float*)d_in[11];
  const float* g1Wr  = (const float*)d_in[12];
  const float* g1Ur  = (const float*)d_in[13];
  const float* g1Wz  = (const float*)d_in[14];
  const float* g1Uz  = (const float*)d_in[15];
  const float* g1Wg  = (const float*)d_in[16];
  const float* g1Ug  = (const float*)d_in[17];
  const float* g1bg  = (const float*)d_in[18];
  const float* ln2g  = (const float*)d_in[19];
  const float* ln2b  = (const float*)d_in[20];
  const float* W1    = (const float*)d_in[21];
  const float* b1    = (const float*)d_in[22];
  const float* W2    = (const float*)d_in[23];
  const float* b2    = (const float*)d_in[24];
  const float* g2Wr  = (const float*)d_in[25];
  const float* g2Ur  = (const float*)d_in[26];
  const float* g2Wz  = (const float*)d_in[27];
  const float* g2Uz  = (const float*)d_in[28];
  const float* g2Wg  = (const float*)d_in[29];
  const float* g2Ug  = (const float*)d_in[30];
  const float* g2bg  = (const float*)d_in[31];
  float* outF = (float*)d_out;

  char* ws = (char*)d_ws;
  const size_t SBsz = (size_t)TOK * 256 * 2;  // 32 MiB bf16 slot
  const size_t LBsz = (size_t)TOK * 512 * 2;  // 64 MiB slot (bf16 512-wide or f32 256-wide)
  u16* slotA = (u16*)(ws);
  u16* slotB = (u16*)(ws + SBsz);
  u16* slotC = (u16*)(ws + 2 * SBsz);
  u16* slotD = (u16*)(ws + 3 * SBsz);
  char* pE = ws + 4 * SBsz;
  char* pF = pE + LBsz;
  char* pG = pF + LBsz;
  char* pH = pG + LBsz;
  float* kvb = (float*)(pH + LBsz);
  float* sb  = (float*)((char*)kvb + 64 * 2048 * 4);
  u16* wpool = (u16*)((char*)sb + 64 * 64 * 4);

  // weight bf16 pool
  u16* wq   = wpool;           u16* wk   = wq + 65536;
  u16* wv   = wk + 65536;      u16* wo   = wv + 65536;
  u16* w1r  = wo + 65536;      u16* w1ur = w1r + 65536;
  u16* w1z  = w1ur + 65536;    u16* w1uz = w1z + 65536;
  u16* w1g  = w1uz + 65536;    u16* w1ug = w1g + 65536;
  u16* wm1  = w1ug + 65536;    u16* wm2  = wm1 + 131072;
  u16* w2r  = wm2 + 131072;    u16* w2ur = w2r + 65536;
  u16* w2z  = w2ur + 65536;    u16* w2uz = w2z + 65536;
  u16* w2g  = w2uz + 65536;    u16* w2ug = w2g + 65536;

  dim3 blk(256);
  auto castW = [&](const float* s, u16* d, int n) {
    cast_kernel<<<(n + 255) / 256, blk, 0, stream>>>(s, d, n);
  };
  // single-input GEMM, bf16 out (+bias, optional relu via template below)
  auto gemmB = [&](const u16* A, const u16* Bw, const float* bias, u16* Cb,
                   int M, int N, int K, float osc) {
    gemm_bf16_kernel<8 | 16, false><<<dim3(N / 128, M / 128), blk, 0, stream>>>(
        A, Bw, nullptr, nullptr, bias, nullptr, Cb, M, N, K, osc);
  };
  auto gemmBR = [&](const u16* A, const u16* Bw, const float* bias, u16* Cb,
                    int M, int N, int K) {
    gemm_bf16_kernel<8 | 16 | 2, false><<<dim3(N / 128, M / 128), blk, 0, stream>>>(
        A, Bw, nullptr, nullptr, bias, nullptr, Cb, M, N, K, 1.f);
  };
  // dual-input GEMM, f32 out:  C = A1@B1 + A2@B2
  auto gemmD = [&](const u16* A1, const u16* B1, const u16* A2, const u16* B2,
                   float* Cf, int M, int N, int K) {
    gemm_bf16_kernel<4, true><<<dim3(N / 128, M / 128), blk, 0, stream>>>(
        A1, B1, A2, B2, nullptr, Cf, nullptr, M, N, K, 1.f);
  };

  castW(Wq, wq, 65536);   castW(Wk, wk, 65536);   castW(Wv, wv, 65536);
  castW(Wo, wo, 65536);
  castW(g1Wr, w1r, 65536); castW(g1Ur, w1ur, 65536);
  castW(g1Wz, w1z, 65536); castW(g1Uz, w1uz, 65536);
  castW(g1Wg, w1g, 65536); castW(g1Ug, w1ug, 65536);
  castW(W1, wm1, 131072);  castW(W2, wm2, 131072);
  castW(g2Wr, w2r, 65536); castW(g2Ur, w2ur, 65536);
  castW(g2Wz, w2z, 65536); castW(g2Uz, w2uz, 65536);
  castW(g2Wg, w2g, 65536); castW(g2Ug, w2ug, 65536);

  // buffer aliases along the dependency chain
  u16* hn = slotA;  u16* qbf = slotB;  u16* kbf = slotC;  u16* vbf = slotD;
  u16* pq = (u16*)pE; u16* pk = (u16*)pF;
  u16* attnb = slotB; u16* ybf = slotC; u16* xbf = slotD;
  float* Rf = (float*)pE; float* Zf = (float*)pF; float* Gf = (float*)pG;
  u16* rxbf = slotA;
  float* out1f = (float*)pH; u16* out1bf = slotA;
  u16* h2n = slotB; u16* tbf = (u16*)pE; u16* y2bf = slotC;
  float* R2f = (float*)pF; float* Z2f = (float*)pG; float* G2f = (float*)pE;
  u16* rx2bf = slotD;

  const float QS = 0.42044820762685725f;  // 32^-0.25 baked into Q/K output
  const int NB = NEL / 256;

  // ---- attention sublayer ----
  ln_kernel<<<TOK, blk, 0, stream>>>(x, ln1g, ln1b, hn);
  gemmB(hn, wq, bq, qbf, TOK, 256, 256, QS);
  gemmB(hn, wk, bk, kbf, TOK, 256, 256, QS);
  gemmB(hn, wv, bv, vbf, TOK, 256, 256, 1.f);
  phi_kernel<<<dim3(TOK / 128, 8), blk, 0, stream>>>(qbf, omega, pq);
  phi_kernel<<<dim3(TOK / 128, 8), blk, 0, stream>>>(kbf, omega, pk);
  kv_kernel<<<64, blk, 0, stream>>>(pk, vbf, kvb, sb);
  num_kernel<<<dim3(64, 64), blk, 0, stream>>>(pq, kvb, sb, attnb);
  gemmB(attnb, wo, bo, ybf, TOK, 256, 256, 1.f);

  // ---- GRU gate 1 (x = input residual, y = attention out) ----
  cast_kernel<<<NB, blk, 0, stream>>>(x, xbf, NEL);
  gemmD(ybf, w1r, xbf, w1ur, Rf, TOK, 256, 256);      // R = y@Wr + x@Ur
  rx_kernel<<<NB, blk, 0, stream>>>(Rf, x, rxbf, NEL);
  gemmD(ybf, w1z, xbf, w1uz, Zf, TOK, 256, 256);      // Z = y@Wz + x@Uz
  gemmD(ybf, w1g, rxbf, w1ug, Gf, TOK, 256, 256);     // G = y@Wg + (r*x)@Ug
  gate_combine_kernel<<<NB, blk, 0, stream>>>(Zf, Gf, x, g1bg, out1f, out1bf, 1, NEL);

  // ---- MLP sublayer ----
  ln_kernel<<<TOK, blk, 0, stream>>>(out1f, ln2g, ln2b, h2n);
  gemmBR(h2n, wm1, b1, tbf, TOK, 512, 256);
  gemmBR(tbf, wm2, b2, y2bf, TOK, 256, 512);

  // ---- GRU gate 2 (x = out1, y = mlp out) ----
  gemmD(y2bf, w2r, out1bf, w2ur, R2f, TOK, 256, 256);
  rx_kernel<<<NB, blk, 0, stream>>>(R2f, out1f, rx2bf, NEL);
  gemmD(y2bf, w2z, out1bf, w2uz, Z2f, TOK, 256, 256);
  gemmD(y2bf, w2g, rx2bf, w2ug, G2f, TOK, 256, 256);
  gate_combine_kernel<<<NB, blk, 0, stream>>>(Z2f, G2f, out1f, g2bg, outF, nullptr, 0, NEL);
}
